// CapsuleNetwork_52587579572429
// MI455X (gfx1250) — matrix operand order repaired and checked by reading, not compiled, not run on hardware
//
#include <hip/hip_runtime.h>

typedef __attribute__((ext_vector_type(16))) __bf16 v16bf;
typedef __attribute__((ext_vector_type(8)))  float  v8f;

#define NUM_NODES 1152
#define BATCH     512
#define KDIM      20736              // 256*81
#define NPIX      (BATCH * 36)       // 18432
#define RELU_ELEMS ((size_t)BATCH * 256 * 400)
#define RELU_BYTES (RELU_ELEMS * 2)          // bf16, 104,857,600 B
#define U_BYTES    ((size_t)BATCH * NUM_NODES * 8 * 4)   // 18,874,368 B
#define WBF_OFF    (RELU_BYTES + U_BYTES)                // 123,731,968 (256-aligned)
#define ROUTE_SMEM ((NUM_NODES * 16 + NUM_NODES + NUM_NODES + 128 * 16) * 4)  // 91136 B

__device__ __forceinline__ unsigned short f2bf(float f) {
    unsigned int u = __float_as_uint(f);
    unsigned int r = ((u >> 16) & 1u) + 0x7FFFu;   // round-to-nearest-even
    return (unsigned short)((u + r) >> 16);
}
__device__ __forceinline__ unsigned int pack2bf(float a, float b) {
    return (unsigned int)f2bf(a) | ((unsigned int)f2bf(b) << 16);
}

// ---------------------------------------------------------------------------
// Kernel 0: one-time pack of prim_w f32 -> bf16 (10.6 MB, stays L2-resident;
// read 288x by the GEMM, so pre-packing halves A-side bytes and makes the
// A tile a pure contiguous copy suitable for async global->LDS DMA).
// ---------------------------------------------------------------------------
__global__ void caps_wcvt_kernel(const float* __restrict__ w,
                                 unsigned short* __restrict__ wbf) {
    const size_t i = ((size_t)blockIdx.x * 256 + threadIdx.x) * 4;  // < 5308416
    const float4 f = *(const float4*)(w + i);
    uint2 o = { pack2bf(f.x, f.y), pack2bf(f.z, f.w) };
    *(uint2*)(wbf + i) = o;
}

// ---------------------------------------------------------------------------
// Kernel 1: Conv2d(1,256,k=9) + bias + ReLU -> bf16 activations [B,256,20,20]
// ---------------------------------------------------------------------------
__global__ void caps_conv1_kernel(const float* __restrict__ img,
                                  const float* __restrict__ w,
                                  const float* __restrict__ bias,
                                  unsigned short* __restrict__ relu_out) {
    __shared__ float simg[784];
    __shared__ float sw[81];
    const int tid = threadIdx.x;             // 256 threads
    const int bc  = blockIdx.x;              // 512*256 blocks
    const int b   = bc >> 8;
    const int ch  = bc & 255;

    for (int i = tid; i < 784; i += 256) simg[i] = img[b * 784 + i];
    if (tid < 81) sw[tid] = w[ch * 81 + tid];
    const float bv = bias[ch];
    __syncthreads();

    const size_t obase = ((size_t)b * 256 + ch) * 400;
    for (int s = tid; s < 400; s += 256) {
        const int oh = s / 20, ow = s % 20;
        float acc = bv;
        #pragma unroll
        for (int kh = 0; kh < 9; kh++)
            #pragma unroll
            for (int kw = 0; kw < 9; kw++)
                acc = fmaf(simg[(oh + kh) * 28 + ow + kw], sw[kh * 9 + kw], acc);
        acc = fmaxf(acc, 0.0f);
        relu_out[obase + s] = f2bf(acc);
    }
}

// ---------------------------------------------------------------------------
// Kernel 2: PrimaryCaps conv as implicit GEMM, v_wmma_f32_16x16x32_bf16.
// M=256, N=18432, K=20736. Block = 4 waves, 64x64 tile, wave = 32x32 sub-tile.
// BK=64, double-buffered LDS, A tile copied with GLOBAL_LOAD_ASYNC_TO_LDS_B128
// (ASYNCcnt) overlapped with WMMA; B tile gathered (im2col) into registers
// during compute and ds_stored at the end of each step. One barrier per step.
// ---------------------------------------------------------------------------
union ABFrag { uint4 u4[2]; v16bf v; };

__global__ __launch_bounds__(128)
void caps_prim_wmma_kernel(const unsigned short* __restrict__ relu,
                           const unsigned short* __restrict__ Wbf,
                           const float* __restrict__ prim_b,
                           float* __restrict__ u) {
    __shared__ uint4 As4[2][512];   // A: 64 rows x 64 k bf16, row-major, x2 buf
    __shared__ uint4 Bs4[2][512];   // B: 64 pix x 64 k bf16, N-major,  x2 buf

    const int tid  = threadIdx.x;        // 0..127
    const int lane = tid & 31;
    const int wave = tid >> 5;
    const int wm   = wave & 1;           // M half of the 64x64 tile
    const int wn   = wave >> 1;          // N half
    const int mbase = blockIdx.y * 64;
    const int pbase = blockIdx.x * 64;

    // B staging: thread -> (pixel, 32-k chunk) gathered from bf16 relu buffer
    const int bpix  = tid >> 1;
    const int bkoff = (tid & 1) * 32;
    const int p  = pbase + bpix;
    const int bb = p / 36;
    const int sp = p - bb * 36;
    const int oh = sp / 6, ow = sp - (sp / 6) * 6;
    const size_t img_base = (size_t)bb * 256 * 400 + (size_t)(oh * 2) * 20 + (ow * 2);

    const int l15 = lane & 15;
    const int kg  = lane >> 4;
    const int ar0 = wm * 32 + l15;        // A rows, sub_m = 0
    const int ar1 = ar0 + 16;             // sub_m = 1
    const int nr0 = wn * 32 + l15;        // B cols, sub_n = 0
    const int nr1 = nr0 + 16;             // sub_n = 1

    v8f acc00 = {}, acc01 = {}, acc10 = {}, acc11 = {};

    // --- helpers -----------------------------------------------------------
    auto issueA = [&](int buf, int kb) {
        #pragma unroll
        for (int i = 0; i < 4; i++) {
            const int q   = tid + 128 * i;   // 0..511 transfers of 16 B
            const int row = q >> 3;
            const int seg = q & 7;
            unsigned lds = (unsigned)(uintptr_t)&As4[buf][row * 8 + seg];
            unsigned long long g = (unsigned long long)(uintptr_t)
                (Wbf + (size_t)(mbase + row) * KDIM + kb + seg * 8);
            asm volatile("global_load_async_to_lds_b128 %0, %1, off"
                         :: "v"(lds), "v"(g) : "memory");
        }
    };
    auto gatherB = [&](int kb, unsigned short us[32]) {
        #pragma unroll
        for (int j = 0; j < 32; j++) {
            const int k  = kb + bkoff + j;
            const int ci = k / 81;
            const int r  = k - ci * 81;
            const int kh = r / 9;
            const int kw = r - kh * 9;
            us[j] = relu[img_base + (size_t)ci * 400 + kh * 20 + kw];
        }
    };
    auto storeB = [&](int buf, const unsigned short us[32]) {
        #pragma unroll
        for (int q = 0; q < 4; q++) {
            uint4 v = { (unsigned)us[8*q+0] | ((unsigned)us[8*q+1] << 16),
                        (unsigned)us[8*q+2] | ((unsigned)us[8*q+3] << 16),
                        (unsigned)us[8*q+4] | ((unsigned)us[8*q+5] << 16),
                        (unsigned)us[8*q+6] | ((unsigned)us[8*q+7] << 16) };
            Bs4[buf][bpix * 8 + (bkoff >> 3) + q] = v;
        }
    };

    // --- prologue: fill buffer 0 ------------------------------------------
    {
        unsigned short us[32];
        issueA(0, 0);
        gatherB(0, us);
        storeB(0, us);
        asm volatile("s_wait_asynccnt 0x0" ::: "memory");
    }
    __syncthreads();

    const int S = KDIM / 64;   // 324 steps
    for (int s = 0; s < S; s++) {
        const int cur = s & 1;
        const int nxt = cur ^ 1;
        unsigned short us[32];

        if (s + 1 < S) {
            issueA(nxt, (s + 1) * 64);     // async DMA into next A buffer
            gatherB((s + 1) * 64, us);     // B gathers in flight during WMMA
        }

        const uint4* Au = As4[cur];
        const uint4* Bu = Bs4[cur];
        #pragma unroll
        for (int cc = 0; cc < 2; cc++) {   // two 32-deep k-chunks
            ABFrag a0, a1, b0, b1;
            a0.u4[0] = Au[ar0 * 8 + cc * 4 + kg];
            a0.u4[1] = Au[ar0 * 8 + cc * 4 + 2 + kg];
            a1.u4[0] = Au[ar1 * 8 + cc * 4 + kg];
            a1.u4[1] = Au[ar1 * 8 + cc * 4 + 2 + kg];
            b0.u4[0] = Bu[nr0 * 8 + cc * 4 + kg];
            b0.u4[1] = Bu[nr0 * 8 + cc * 4 + 2 + kg];
            b1.u4[0] = Bu[nr1 * 8 + cc * 4 + kg];
            b1.u4[1] = Bu[nr1 * 8 + cc * 4 + 2 + kg];

            acc00 = __builtin_amdgcn_wmma_f32_16x16x32_bf16(false, a0.v, false, b0.v,
                                                            (short)0, acc00, false, false);
            acc01 = __builtin_amdgcn_wmma_f32_16x16x32_bf16(false, a0.v, false, b1.v,
                                                            (short)0, acc01, false, false);
            acc10 = __builtin_amdgcn_wmma_f32_16x16x32_bf16(false, a1.v, false, b0.v,
                                                            (short)0, acc10, false, false);
            acc11 = __builtin_amdgcn_wmma_f32_16x16x32_bf16(false, a1.v, false, b1.v,
                                                            (short)0, acc11, false, false);
        }

        if (s + 1 < S) storeB(nxt, us);
        asm volatile("s_wait_asynccnt 0x0" ::: "memory");
        __syncthreads();
    }

    // ---- store: D layout lane=N, VGPR r -> M = r + 8*(lane>>4) ----
    v8f accs[4] = { acc00, acc01, acc10, acc11 };
    #pragma unroll
    for (int sm = 0; sm < 2; sm++) {
        #pragma unroll
        for (int sn = 0; sn < 2; sn++) {
            const v8f a = accs[sm * 2 + sn];
            const int pg = pbase + wn * 32 + sn * 16 + l15;   // global pixel
            const int ob = pg / 36;
            const int os = pg - ob * 36;
            #pragma unroll
            for (int r = 0; r < 8; r++) {
                const int gm = mbase + wm * 32 + sm * 16 + kg * 8 + r;  // channel
                const int d  = gm >> 5;       // capsule dim 0..7
                const int g  = gm & 31;       // capsule group 0..31
                u[((size_t)ob * NUM_NODES + g * 36 + os) * 8 + d] = a[r] + prim_b[gm];
            }
        }
    }
}

// ---------------------------------------------------------------------------
// Kernel 3: squash over the 8-dim capsules, in place. One thread per (b,node).
// ---------------------------------------------------------------------------
__global__ void caps_squash_kernel(float* __restrict__ u) {
    const size_t idx = (size_t)blockIdx.x * 256 + threadIdx.x;  // < 512*1152
    float* p = u + idx * 8;
    float v[8], sn = 0.0f;
    #pragma unroll
    for (int i = 0; i < 8; i++) { v[i] = p[i]; sn = fmaf(v[i], v[i], sn); }
    const float scale = sn / ((1.0f + sn) * sqrtf(sn));
    #pragma unroll
    for (int i = 0; i < 8; i++) p[i] = v[i] * scale;
}

// ---------------------------------------------------------------------------
// Kernel 4: fused DigitCaps + 3 routing iterations. One block per (c,b).
// u_hat (1152x16 f32) lives in LDS for all iterations: the 377 MB u_hat
// tensor never touches HBM. W slice stays L2-resident across batch blocks.
// ---------------------------------------------------------------------------
__global__ __launch_bounds__(128)
void caps_routing_kernel(const float* __restrict__ u,
                         const float* __restrict__ W,
                         float* __restrict__ out) {
    extern __shared__ float smem[];
    float* uhat = smem;                       // 1152*16
    float* bij  = uhat + NUM_NODES * 16;      // 1152
    float* cv   = bij  + NUM_NODES;           // 1152
    float* red  = cv   + NUM_NODES;           // 128*16

    const int tid = threadIdx.x;              // 128 threads
    const int cb  = blockIdx.x;               // 5120 blocks
    const int c   = cb % 10;
    const int b   = cb / 10;

    const float* ub = u + (size_t)b * NUM_NODES * 8;
    const float* Wc = W + (size_t)c * NUM_NODES * 128;

    #pragma unroll
    for (int t = 0; t < 9; t++) {
        const int n = tid + t * 128;
        const float* un = ub + (size_t)n * 8;
        const float* Wn = Wc + (size_t)n * 128;
        float uu[8];
        #pragma unroll
        for (int i = 0; i < 8; i++) uu[i] = un[i];
        float o16[16];
        #pragma unroll
        for (int o = 0; o < 16; o++) o16[o] = 0.0f;
        #pragma unroll
        for (int i = 0; i < 8; i++)
            #pragma unroll
            for (int o = 0; o < 16; o++)
                o16[o] = fmaf(uu[i], Wn[i * 16 + o], o16[o]);
        #pragma unroll
        for (int o = 0; o < 16; o++) uhat[n * 16 + o] = o16[o];
        bij[n] = 0.0f;
    }
    __syncthreads();

    for (int iter = 0; iter < 3; iter++) {
        float mx = -3.402823466e+38f;
        #pragma unroll
        for (int t = 0; t < 9; t++) mx = fmaxf(mx, bij[tid + t * 128]);
        red[tid] = mx;
        __syncthreads();
        for (int off = 64; off > 0; off >>= 1) {
            if (tid < off) red[tid] = fmaxf(red[tid], red[tid + off]);
            __syncthreads();
        }
        mx = red[0];
        __syncthreads();

        float psum = 0.0f;
        #pragma unroll
        for (int t = 0; t < 9; t++) {
            const int n = tid + t * 128;
            const float e = expf(bij[n] - mx);
            cv[n] = e;
            psum += e;
        }
        red[tid] = psum;
        __syncthreads();
        for (int off = 64; off > 0; off >>= 1) {
            if (tid < off) red[tid] += red[tid + off];
            __syncthreads();
        }
        const float inv = 1.0f / red[0];
        __syncthreads();

        float sp[16];
        #pragma unroll
        for (int o = 0; o < 16; o++) sp[o] = 0.0f;
        #pragma unroll
        for (int t = 0; t < 9; t++) {
            const int n = tid + t * 128;
            const float cc = cv[n] * inv;
            #pragma unroll
            for (int o = 0; o < 16; o++)
                sp[o] = fmaf(cc, uhat[n * 16 + o], sp[o]);
        }
        #pragma unroll
        for (int o = 0; o < 16; o++) red[tid * 16 + o] = sp[o];
        __syncthreads();
        for (int off = 64; off > 0; off >>= 1) {
            if (tid < off)
                #pragma unroll
                for (int o = 0; o < 16; o++)
                    red[tid * 16 + o] += red[(tid + off) * 16 + o];
            __syncthreads();
        }
        float s[16], sn = 0.0f;
        #pragma unroll
        for (int o = 0; o < 16; o++) { s[o] = red[o]; sn = fmaf(s[o], s[o], sn); }
        __syncthreads();
        const float scale = sn / ((1.0f + sn) * sqrtf(sn));
        float v[16];
        #pragma unroll
        for (int o = 0; o < 16; o++) v[o] = scale * s[o];

        if (iter < 2) {
            #pragma unroll
            for (int t = 0; t < 9; t++) {
                const int n = tid + t * 128;
                float a = 0.0f;
                #pragma unroll
                for (int o = 0; o < 16; o++)
                    a = fmaf(uhat[n * 16 + o], v[o], a);
                bij[n] += a;
            }
            __syncthreads();
        } else {
            if (tid < 16) out[((size_t)b * 10 + c) * 16 + tid] = v[tid];
        }
    }
}

// ---------------------------------------------------------------------------
extern "C" void kernel_launch(void* const* d_in, const int* in_sizes, int n_in,
                              void* d_out, int out_size, void* d_ws, size_t ws_size,
                              hipStream_t stream) {
    const float* images  = (const float*)d_in[0];
    const float* conv1_w = (const float*)d_in[1];
    const float* conv1_b = (const float*)d_in[2];
    const float* prim_w  = (const float*)d_in[3];
    const float* prim_b  = (const float*)d_in[4];
    const float* W       = (const float*)d_in[5];
    float* out = (float*)d_out;

    char* ws = (char*)d_ws;
    unsigned short* relu = (unsigned short*)ws;              // 104.9 MB bf16
    float*          u    = (float*)(ws + RELU_BYTES);        // 18.9 MB f32
    unsigned short* wbf  = (unsigned short*)(ws + WBF_OFF);  // 10.6 MB bf16

    // 0) pack PrimaryCaps weights to bf16 (5,308,416 elems, 4 per thread)
    caps_wcvt_kernel<<<5184, 256, 0, stream>>>(prim_w, wbf);

    // 1) Conv1 + ReLU -> bf16
    caps_conv1_kernel<<<512 * 256, 256, 0, stream>>>(images, conv1_w, conv1_b, relu);

    // 2) PrimaryCaps implicit GEMM (WMMA bf16 + async A-tile DMA) -> u[b][n][i]
    dim3 g2(NPIX / 64, 256 / 64);
    caps_prim_wmma_kernel<<<g2, 128, 0, stream>>>(relu, wbf, prim_b, u);

    // 3) squash
    caps_squash_kernel<<<(512 * NUM_NODES) / 256, 256, 0, stream>>>(u);

    // 4) fused DigitCaps + routing (91,136 B dynamic LDS per block)
    (void)hipFuncSetAttribute((const void*)caps_routing_kernel,
                              hipFuncAttributeMaxDynamicSharedMemorySize, ROUTE_SMEM);
    caps_routing_kernel<<<10 * 512, 128, ROUTE_SMEM, stream>>>(u, W, out);
}
